// ConvSPE_75557064671917
// MI455X (gfx1250) — matrix operand order repaired and checked by reading, not compiled, not run on hardware
//
#include <hip/hip_runtime.h>

// ---------------- problem constants (from reference) ----------------
#define B_  2
#define S_  1024
#define H_  8
#define D_  64
#define R_  32
#define K_  128
#define T_  (4*K_ + S_)         // 1536, z time length
#define C_  (H_*D_)             // 512 channels
#define NOUT1 (B_*S_*H_*R_)     // 524288 elements per output tensor

// ---------------- tiling ----------------
#define SCHUNK 128              // s values per block (8 waves x 16)
#define WIN    272              // z time window per d: (SCHUNK-16) + 160
#define ZSTR   280              // padded LDS row stride (halves) -> conflict-free
#define DSPLIT 2                // split contraction over d across blocks
#define DPB    (D_/DSPLIT)      // 32 d per block
#define NKC    5                // k-chunks of 32 over j' in [0,160)
#define WPAD   176              // padded weight row: 16 front + 128 + 32 back

static_assert(WIN == (SCHUNK - 16) + NKC*32, "window");

typedef __attribute__((ext_vector_type(16))) _Float16     v16h;
typedef __attribute__((ext_vector_type(8)))  float        v8f;
typedef __attribute__((ext_vector_type(4)))  float        v4f;
typedef __attribute__((ext_vector_type(4)))  unsigned int v4u;
typedef __attribute__((ext_vector_type(2)))  unsigned int v2u;

union AFrag { v16h v; v4u q[2]; };          // 16-bit A: two 16B runs per lane
union BFrag { v16h v; unsigned int w[8]; }; // 16-bit B: two 8-K runs per lane, same K order as A

__global__ __launch_bounds__(256)
void conv_spe_wmma(const float* __restrict__ qry,
                   const float* __restrict__ key,
                   const float* __restrict__ wq,
                   const float* __restrict__ wk,
                   const float* __restrict__ z,
                   float* __restrict__ part)
{
    // LDS: double-buffered f16 z window + per-d padded conv weights (f32)
    __shared__ unsigned short zbuf[2][32 * ZSTR];          // ~35.8 KB
    __shared__ float          wbuf[2][2][WPAD];            // ~2.8 KB

    const int tid = threadIdx.x;
    int blk = blockIdx.x;
    const int ds = blk & 1;  blk >>= 1;        // d-split index
    const int sc = blk & 7;  blk >>= 3;        // s-chunk
    const int h  = blk & 7;  blk >>= 3;        // head
    const int b  = blk;                        // batch
    const int s0 = sc * SCHUNK;
    const int d0 = ds * DPB;
    const int tbase = K_ + s0;                 // z time offset of window start

    const int lane = tid & 31;
    const int wid  = tid >> 5;                 // wave = one 16-wide s tile
    const int n    = lane & 15;                // s column within tile
    const int hi   = lane >> 4;                // lane half
    const int sg   = s0 + wid * 16 + n;        // global s for this lane's column

    // stage z window (f32 -> f16) and padded weights for channel d
    auto stage = [&](int dloc, int buf) {
        const int dd = d0 + dloc;
        const size_t chan = (size_t)(h * D_ + dd) * T_ + tbase;
        for (int i = tid; i < 32 * (WIN / 4); i += 256) {
            const int r  = i / (WIN / 4);
            const int c4 = i - r * (WIN / 4);
            const float* gp = z + (size_t)(b * R_ + r) * ((size_t)C_ * T_) + chan + c4 * 4;
            if (c4 == 0)  // warm next channel's row (speculative, drop-safe)
                __builtin_prefetch(gp + (size_t)T_, 0, 1);
            v4f x = *(const v4f*)gp;
            unsigned int lo = __builtin_bit_cast(unsigned int, __builtin_amdgcn_cvt_pkrtz(x.x, x.y));
            unsigned int hh = __builtin_bit_cast(unsigned int, __builtin_amdgcn_cvt_pkrtz(x.z, x.w));
            v2u t; t.x = lo; t.y = hh;
            *(v2u*)&zbuf[buf][r * ZSTR + c4 * 4] = t;
        }
        // set 0 = wk (drives qhat, per the reference's swap), set 1 = wq (drives khat)
        for (int i = tid; i < 2 * WPAD; i += 256) {
            const int set = i / WPAD;
            const int p   = i - set * WPAD;
            const int j   = p - 16;
            const float* wsrc = set ? wq : wk;
            wbuf[buf][set][p] =
                ((unsigned)j < (unsigned)K_) ? wsrc[(h * D_ + dd) * K_ + j] : 0.0f;
        }
    };

    v8f mq0 = {}, mq1 = {}, mk0 = {}, mk1 = {};   // main f32 accumulators

    stage(0, 0);
    __syncthreads();

    for (int dloc = 0; dloc < DPB; ++dloc) {
        const int buf = dloc & 1;
        if (dloc + 1 < DPB) stage(dloc + 1, buf ^ 1);

        const int dd = d0 + dloc;
        const float qv = qry[((size_t)(b * S_ + sg) * H_ + h) * D_ + dd];
        const float kv = key[((size_t)(b * S_ + sg) * H_ + h) * D_ + dd];

        const unsigned short* zb  = &zbuf[buf][0];
        const float*          wbq = &wbuf[buf][0][0];
        const float*          wbk = &wbuf[buf][1][0];

        v8f iq0 = {}, iq1 = {}, ik0 = {}, ik1 = {};   // per-d inner accumulators

#pragma unroll
        for (int ch = 0; ch < NKC; ++ch) {
            const int jc = ch * 32;

            // ---- A fragments: z window, documented 16-bit A layout ----
            // lanes 0-15: K={0..7,16..23}; lanes 16-31: K={8..15,24..31}
            const int klo = hi * 8;
            const int tA  = wid * 16 + jc + klo;      // 16B aligned (klo*2 in {0,16})
            AFrag a0, a1;
            {
                const unsigned short* p0 = zb + (size_t)n * ZSTR + tA;          // rows 0..15
                a0.q[0] = *(const v4u*)p0;
                a0.q[1] = *(const v4u*)(p0 + 16);
                const unsigned short* p1 = zb + (size_t)(16 + n) * ZSTR + tA;   // rows 16..31
                a1.q[0] = *(const v4u*)p1;
                a1.q[1] = *(const v4u*)(p1 + 16);
            }

            // ---- B fragments: padded Toeplitz of conv weights ----
            // lanes 0-15: K={0..7,16..23}; lanes 16-31: K={8..15,24..31}
            const int p0 = 16 + jc + hi * 8 - n;      // in [1, 152]; +23 <= 175 < WPAD
            BFrag bq, bk;
#pragma unroll
            for (int e = 0; e < 8; ++e) {
                const int o = p0 + 2 * e + (e >= 4 ? 8 : 0);
                bq.w[e] = __builtin_bit_cast(unsigned int,
                    __builtin_amdgcn_cvt_pkrtz(wbq[o], wbq[o + 1]));
                bk.w[e] = __builtin_bit_cast(unsigned int,
                    __builtin_amdgcn_cvt_pkrtz(wbk[o], wbk[o + 1]));
            }

            iq0 = __builtin_amdgcn_wmma_f32_16x16x32_f16(false, a0.v, false, bq.v, (short)0, iq0, false, false);
            iq1 = __builtin_amdgcn_wmma_f32_16x16x32_f16(false, a1.v, false, bq.v, (short)0, iq1, false, false);
            ik0 = __builtin_amdgcn_wmma_f32_16x16x32_f16(false, a0.v, false, bk.v, (short)0, ik0, false, false);
            ik1 = __builtin_amdgcn_wmma_f32_16x16x32_f16(false, a1.v, false, bk.v, (short)0, ik1, false, false);
        }

        // per-d post-scale: C column s == lane's n -> lane-uniform scalar FMA
#pragma unroll
        for (int v = 0; v < 8; ++v) {
            mq0[v] += qv * iq0[v];
            mq1[v] += qv * iq1[v];
            mk0[v] += kv * ik0[v];
            mk1[v] += kv * ik1[v];
        }
        __syncthreads();
    }

    // ---- store partials: out[b,s,h,r], lane holds 8 consecutive r ----
    float* pq = part + (size_t)ds * (2 * (size_t)NOUT1)
                     + ((size_t)(b * S_ + sg) * H_ + h) * R_;
    float* pk = pq + NOUT1;
    const int r0 = hi * 8;
    v4f t;
    t.x=mq0[0]; t.y=mq0[1]; t.z=mq0[2]; t.w=mq0[3]; *(v4f*)(pq + r0)          = t;
    t.x=mq0[4]; t.y=mq0[5]; t.z=mq0[6]; t.w=mq0[7]; *(v4f*)(pq + r0 + 4)      = t;
    t.x=mq1[0]; t.y=mq1[1]; t.z=mq1[2]; t.w=mq1[3]; *(v4f*)(pq + 16 + r0)     = t;
    t.x=mq1[4]; t.y=mq1[5]; t.z=mq1[6]; t.w=mq1[7]; *(v4f*)(pq + 16 + r0 + 4) = t;
    t.x=mk0[0]; t.y=mk0[1]; t.z=mk0[2]; t.w=mk0[3]; *(v4f*)(pk + r0)          = t;
    t.x=mk0[4]; t.y=mk0[5]; t.z=mk0[6]; t.w=mk0[7]; *(v4f*)(pk + r0 + 4)      = t;
    t.x=mk1[0]; t.y=mk1[1]; t.z=mk1[2]; t.w=mk1[3]; *(v4f*)(pk + 16 + r0)     = t;
    t.x=mk1[4]; t.y=mk1[5]; t.z=mk1[6]; t.w=mk1[7]; *(v4f*)(pk + 16 + r0 + 4) = t;
}

// deterministic d-split reduction (no FP atomics)
__global__ __launch_bounds__(256)
void reduce_parts(const float* __restrict__ part, float* __restrict__ out)
{
    const size_t i = (size_t)blockIdx.x * 256 + threadIdx.x;
    if (i < (size_t)2 * NOUT1)
        out[i] = part[i] + part[i + (size_t)2 * NOUT1];
}

extern "C" void kernel_launch(void* const* d_in, const int* in_sizes, int n_in,
                              void* d_out, int out_size, void* d_ws, size_t ws_size,
                              hipStream_t stream)
{
    const float* qry = (const float*)d_in[0];
    const float* key = (const float*)d_in[1];
    const float* wq  = (const float*)d_in[2];
    const float* wk  = (const float*)d_in[3];
    const float* z   = (const float*)d_in[4];
    float* part = (float*)d_ws;                 // 2 x 2*NOUT1 f32 = 8 MB
    float* out  = (float*)d_out;

    conv_spe_wmma<<<dim3(B_ * H_ * (S_ / SCHUNK) * DSPLIT), dim3(256), 0, stream>>>(
        qry, key, wq, wk, z, part);
    reduce_parts<<<dim3((2 * NOUT1 + 255) / 256), dim3(256), 0, stream>>>(part, out);
}
